// NodeEdgeAggregatorV4_58944131170470
// MI455X (gfx1250) — matrix operand order repaired and checked by reading, not compiled, not run on hardware
//
#include <hip/hip_runtime.h>
#include <math.h>

#define LRELU_NEG 0.2f

typedef __attribute__((ext_vector_type(16))) __bf16 v16bf;
typedef __attribute__((ext_vector_type(2)))  __bf16 v2bf;
typedef __attribute__((ext_vector_type(8)))  float  v8f;

__device__ __forceinline__ v16bf pack16(const float* f) {
  union { v16bf v; v2bf p[8]; } u;
#if __has_builtin(__builtin_amdgcn_cvt_pk_bf16_f32)
#pragma unroll
  for (int i = 0; i < 8; ++i) u.p[i] = __builtin_amdgcn_cvt_pk_bf16_f32(f[2 * i], f[2 * i + 1]);
#else
  // standard fptrunc (RNE); backend selects native bf16 cvt when available
#pragma unroll
  for (int i = 0; i < 16; ++i) u.v[i] = (__bf16)f[i];
#endif
  return u.v;
}

__device__ __forceinline__ float lrelu(float v) { return v > 0.f ? v : LRELU_NEG * v; }

__device__ __forceinline__ void atomicMaxF(float* addr, float v) {
  // ordered-float trick; works for mixed signs when initialized with a float
  if (v >= 0.f) atomicMax((int*)addr, __float_as_int(v));
  else          atomicMin((unsigned int*)addr, __float_as_uint(v));
}

// ---------------- WMMA GEMM: D = act(A1@B1 + rowscale(A2)@B2 + crowscale*Cadd) --------
// A: [M,K] row-major fp32 (converted to bf16 on load), B: [K, NCT*16] fp32 weights
// staged to LDS in v_wmma_f32_16x16x32_bf16 fragment layout.

__device__ __forceinline__ v16bf load_a_frag(const float* __restrict__ A, int lda,
                                             int row, int kb, int K, float scale) {
  // 16-bit A-matrix 16x32 layout: per lane K = {kb..kb+7, kb+16..kb+23}
  // (kb already lane-adjusted: +8 for lanes 16-31).
  const float* p = A + (size_t)row * lda;
  float f[16];
  if (kb + 24 <= K) {  // uniform fast path (K multiple of 32): 4x b128 loads
    float4 x0 = *(const float4*)(p + kb);
    float4 x1 = *(const float4*)(p + kb + 4);
    float4 y0 = *(const float4*)(p + kb + 16);
    float4 y1 = *(const float4*)(p + kb + 20);
    f[0] = x0.x;  f[1] = x0.y;  f[2]  = x0.z;  f[3]  = x0.w;
    f[4] = x1.x;  f[5] = x1.y;  f[6]  = x1.z;  f[7]  = x1.w;
    f[8] = y0.x;  f[9] = y0.y;  f[10] = y0.z;  f[11] = y0.w;
    f[12] = y1.x; f[13] = y1.y; f[14] = y1.z;  f[15] = y1.w;
  } else {           // tail path (only the K=16 time-embed GEMM)
#pragma unroll
    for (int j = 0; j < 8; ++j) {
      int ka = kb + j, kc = kb + 16 + j;
      f[j]     = ka < K ? p[ka] : 0.f;
      f[8 + j] = kc < K ? p[kc] : 0.f;
    }
  }
#pragma unroll
  for (int i = 0; i < 16; ++i) f[i] *= scale;  // scale==1.0f folds away
  return pack16(f);
}

__device__ __forceinline__ void stage_B(const float* __restrict__ B, int K, int N,
                                        __bf16* Bs, int tid) {
  // B fragment layout (32x16, 16-bit): lane holds col N=lane&15;
  // per lane K = {kb..kb+7, kb+16..kb+23}, kb = k0 (+8 for lanes 16-31).
  int nkt = (K + 31) >> 5, nct = N >> 4;
  int total = nkt * nct * 32;
  for (int idx = tid; idx < total; idx += 256) {
    int l  = idx & 31;
    int f  = idx >> 5;
    int ct = f % nct, kt = f / nct;
    int col = (ct << 4) + (l & 15);
    int kb  = (kt << 5) + ((l >> 4) << 3);
    float tmp[16];
#pragma unroll
    for (int e = 0; e < 16; ++e) {
      int k = kb + (e < 8 ? e : e + 8);
      tmp[e] = k < K ? B[(size_t)k * N + col] : 0.f;
    }
    *reinterpret_cast<v16bf*>(Bs + (size_t)idx * 16) = pack16(tmp);
  }
}

template <int NCT>
__global__ __launch_bounds__(256) void gemm2_wmma(
    const float* __restrict__ A1, const float* __restrict__ B1, int K1,
    const float* __restrict__ A2, const float* __restrict__ B2, int K2,
    const float* __restrict__ a2_rowscale,
    const float* __restrict__ Cadd, const float* __restrict__ c_rowscale,
    float* __restrict__ D, int M, int act) {
  constexpr int Nc = NCT * 16;
  __shared__ alignas(32) __bf16 Bs[8 * 8 * 32 * 16];  // 64 KB: up to K=256, N=128
  const int tid  = threadIdx.x;
  const int lane = tid & 31;
  const int wave = tid >> 5;
  const int row0 = blockIdx.x * 128 + wave * 16;
  const bool active = row0 < M;           // M is a multiple of 16; whole waves drop out
  const int lrow  = row0 + (lane & 15);
  const int kboff = (lane >> 4) << 3;     // +8 for upper half-wave (A frag layout)

  v8f acc[NCT];
#pragma unroll
  for (int ct = 0; ct < NCT; ++ct)
#pragma unroll
    for (int r = 0; r < 8; ++r) acc[ct][r] = 0.f;

  stage_B(B1, K1, Nc, Bs, tid);
  __syncthreads();
  if (active) {
    for (int k0 = 0; k0 < K1; k0 += 32) {
      v16bf a = load_a_frag(A1, K1, lrow, k0 + kboff, K1, 1.0f);
      const __bf16* bbase = Bs + (size_t)(k0 >> 5) * (NCT * 512) + (size_t)lane * 16;
      v16bf b[NCT];
#pragma unroll
      for (int ct = 0; ct < NCT; ++ct)
        b[ct] = *reinterpret_cast<const v16bf*>(bbase + (size_t)ct * 512);
#pragma unroll
      for (int ct = 0; ct < NCT; ++ct)
        acc[ct] = __builtin_amdgcn_wmma_f32_16x16x32_bf16(false, a, false, b[ct], (short)0,
                                                          acc[ct], false, false);
    }
  }
  if (A2) {
    __syncthreads();
    stage_B(B2, K2, Nc, Bs, tid);
    __syncthreads();
    if (active) {
      const float sc = a2_rowscale ? a2_rowscale[lrow] : 1.0f;
      for (int k0 = 0; k0 < K2; k0 += 32) {
        v16bf a = load_a_frag(A2, K2, lrow, k0 + kboff, K2, sc);
        const __bf16* bbase = Bs + (size_t)(k0 >> 5) * (NCT * 512) + (size_t)lane * 16;
        v16bf b[NCT];
#pragma unroll
        for (int ct = 0; ct < NCT; ++ct)
          b[ct] = *reinterpret_cast<const v16bf*>(bbase + (size_t)ct * 512);
#pragma unroll
        for (int ct = 0; ct < NCT; ++ct)
          acc[ct] = __builtin_amdgcn_wmma_f32_16x16x32_bf16(false, a, false, b[ct], (short)0,
                                                            acc[ct], false, false);
      }
    }
  }
  if (active) {
    const int colb  = lane & 15;
    const int rbase = row0 + ((lane & 16) >> 1);  // lanes 16-31 -> rows +8
#pragma unroll
    for (int ct = 0; ct < NCT; ++ct) {
      const int col = (ct << 4) + colb;
#pragma unroll
      for (int r = 0; r < 8; ++r) {
        float v = acc[ct][r];
        const int rr = rbase + r;
        if (Cadd) {
          float cs = c_rowscale ? c_rowscale[rr] : 1.0f;
          v += Cadd[(size_t)rr * Nc + col] * cs;
        }
        if (act == 1)      v = v > 0.f ? v : 0.f;
        else if (act == 2) v = lrelu(v);
        D[(size_t)rr * Nc + col] = v;
      }
    }
  }
}

// ---------------- irregular / elementwise kernels ----------------

__global__ void fill_kernel(float* __restrict__ p, float v, long long n) {
  long long i = blockIdx.x * (long long)blockDim.x + threadIdx.x;
  long long s = gridDim.x * (long long)blockDim.x;
  for (; i < n; i += s) p[i] = v;
}

// per-edge scalars: hs = h_row . a_src, hd = h_row . a_dst  (HID=128, wave per edge)
__global__ void edge_score_kernel(const float* __restrict__ h, const float* __restrict__ a_src,
                                  const float* __restrict__ a_dst, float* __restrict__ hs,
                                  float* __restrict__ hd, int E_) {
  int e = blockIdx.x * 8 + (threadIdx.x >> 5);
  if (e >= E_) return;
  int lane = threadIdx.x & 31;
  const float* hp = h + (size_t)e * 128;
  float ss = 0.f, sd = 0.f;
#pragma unroll
  for (int j = 0; j < 4; ++j) {
    int c = lane + 32 * j;
    float v = hp[c];
    ss += v * a_src[c];
    sd += v * a_dst[c];
  }
#pragma unroll
  for (int o = 16; o; o >>= 1) { ss += __shfl_xor(ss, o); sd += __shfl_xor(sd, o); }
  if (lane == 0) { hs[e] = ss; hd[e] = sd; }
}

__global__ void gat_max_kernel(const float* __restrict__ hs, const float* __restrict__ hd,
                               const int* __restrict__ lsrc, const int* __restrict__ ldst,
                               float* __restrict__ mmax, int ELG_) {
  int i = blockIdx.x * blockDim.x + threadIdx.x;
  if (i >= ELG_) return;
  float lg = lrelu(hs[lsrc[i]] + hd[ldst[i]]);
  atomicMaxF(&mmax[ldst[i]], lg);
}

// wave per line-graph edge: den[dst]+=ex ; gat[dst][:] += ex*h[src][:]
__global__ void gat_accum_kernel(const float* __restrict__ h, const float* __restrict__ hs,
                                 const float* __restrict__ hd, const int* __restrict__ lsrc,
                                 const int* __restrict__ ldst, const float* __restrict__ mmax,
                                 float* __restrict__ den, float* __restrict__ gat, int ELG_) {
  int i = blockIdx.x * 8 + (threadIdx.x >> 5);
  if (i >= ELG_) return;
  int lane = threadIdx.x & 31;
  int s = lsrc[i], d = ldst[i];
  float lg = lrelu(hs[s] + hd[d]);
  float ex = __expf(lg - mmax[d]);
  if (lane == 0) atomicAdd(&den[d], ex);
  const float* hp = h + (size_t)s * 128;
  float* gp = gat + (size_t)d * 128;
#pragma unroll
  for (int j = 0; j < 4; ++j) {
    int c = lane + 32 * j;
    atomicAdd(gp + c, ex * hp[c]);
  }
}

__global__ void recip_eps_kernel(float* __restrict__ p, long long n) {
  long long i = blockIdx.x * (long long)blockDim.x + threadIdx.x;
  long long s = gridDim.x * (long long)blockDim.x;
  for (; i < n; i += s) p[i] = 1.f / fmaxf(p[i], 1e-16f);
}

__global__ void recip_max1_kernel(float* __restrict__ p, long long n) {
  long long i = blockIdx.x * (long long)blockDim.x + threadIdx.x;
  long long s = gridDim.x * (long long)blockDim.x;
  for (; i < n; i += s) p[i] = 1.f / fmaxf(p[i], 1.f);
}

// EdgeToNode: scatter edge features to both endpoints (+ incidence counts)
__global__ void scatter_etn_kernel(const float* __restrict__ ef, const int* __restrict__ H0,
                                   const int* __restrict__ H1, float* __restrict__ nacc,
                                   float* __restrict__ cnt, int E_) {
  int e = blockIdx.x * 8 + (threadIdx.x >> 5);
  if (e >= E_) return;
  int lane = threadIdx.x & 31;
  int n0 = H0[e], n1 = H1[e];
  const float* p = ef + (size_t)e * 128;
  float* a0 = nacc + (size_t)n0 * 128;
  float* a1 = nacc + (size_t)n1 * 128;
#pragma unroll
  for (int j = 0; j < 4; ++j) {
    int c = lane + 32 * j;
    float v = p[c];
    atomicAdd(a0 + c, v);
    atomicAdd(a1 + c, v);
  }
  if (lane == 0) { atomicAdd(cnt + n0, 1.f); atomicAdd(cnt + n1, 1.f); }
}

__global__ void etn_finalize_kernel(const float* __restrict__ nacc, const float* __restrict__ cnt,
                                    float* __restrict__ outp, long long total) {
  long long i = blockIdx.x * (long long)blockDim.x + threadIdx.x;
  long long s = gridDim.x * (long long)blockDim.x;
  for (; i < total; i += s) {
    long long n = i >> 7;
    outp[i] = lrelu(nacc[i] / fmaxf(cnt[n], 1.f));
  }
}

__global__ void deg_count_kernel(const int* __restrict__ dst, float* __restrict__ deg, int E_) {
  int i = blockIdx.x * blockDim.x + threadIdx.x;
  if (i >= E_) return;
  atomicAdd(&deg[dst[i]], 1.f);
}

// generic row scatter for SAGE mean aggregation: acc[dst[e]][:] += h[src[e]][:]
__global__ void scatter_rows_kernel(const float* __restrict__ h, const int* __restrict__ src,
                                    const int* __restrict__ dst, float* __restrict__ acc,
                                    int E_, int K) {
  int e = blockIdx.x * 8 + (threadIdx.x >> 5);
  if (e >= E_) return;
  int lane = threadIdx.x & 31;
  const float* hp = h + (size_t)src[e] * K;
  float* ap = acc + (size_t)dst[e] * K;
  for (int c = lane; c < K; c += 32) atomicAdd(ap + c, hp[c]);
}

// MixAttention: wave per node
__global__ void mix_kernel(const float* __restrict__ hn, const float* __restrict__ he,
                           const float* __restrict__ an, const float* __restrict__ ae,
                           float* __restrict__ outm, int Nn) {
  int n = blockIdx.x * 8 + (threadIdx.x >> 5);
  if (n >= Nn) return;
  int lane = threadIdx.x & 31;
  const float* pn = hn + (size_t)n * 128;
  const float* pe = he + (size_t)n * 128;
  float sn = 0.f, se = 0.f;
  float vn[4], ve[4];
#pragma unroll
  for (int j = 0; j < 4; ++j) {
    int c = lane + 32 * j;
    vn[j] = pn[c]; ve[j] = pe[c];
    sn += vn[j] * an[c]; se += ve[j] * ae[c];
  }
#pragma unroll
  for (int o = 16; o; o >>= 1) { sn += __shfl_xor(sn, o); se += __shfl_xor(se, o); }
  sn = lrelu(sn); se = lrelu(se);
  float sm = fmaxf(sn, se);
  float en = __expf(sn - sm), ee = __expf(se - sm);
  float inv = 1.f / (en + ee);
  float* po = outm + (size_t)n * 128;
#pragma unroll
  for (int j = 0; j < 4; ++j) {
    int c = lane + 32 * j;
    po[c] = (en * vn[j] + ee * ve[j]) * inv;
  }
}

// in-place row log_softmax over 64 columns (wave per node, 2 cols/lane)
__global__ void logsoftmax_kernel(float* __restrict__ out, int Nn) {
  int n = blockIdx.x * 8 + (threadIdx.x >> 5);
  if (n >= Nn) return;
  int lane = threadIdx.x & 31;
  float* p = out + (size_t)n * 64;
  float z0 = p[lane], z1 = p[lane + 32];
  float m = fmaxf(z0, z1);
#pragma unroll
  for (int o = 16; o; o >>= 1) m = fmaxf(m, __shfl_xor(m, o));
  float s = __expf(z0 - m) + __expf(z1 - m);
#pragma unroll
  for (int o = 16; o; o >>= 1) s += __shfl_xor(s, o);
  float l = m + __logf(s);
  p[lane] = z0 - l;
  p[lane + 32] = z1 - l;
}

// ---------------- orchestration ----------------

extern "C" void kernel_launch(void* const* d_in, const int* in_sizes, int n_in,
                              void* d_out, int out_size, void* d_ws, size_t ws_size,
                              hipStream_t stream) {
  (void)in_sizes; (void)n_in; (void)out_size; (void)ws_size;
  const int NN = 100000, EE = 500000, LGE = 1000000;
  const int HID = 128, FIN = 256, TD = 16, AD = 32;

  const float* x     = (const float*)d_in[0];
  const float* et    = (const float*)d_in[1];
  const float* ea    = (const float*)d_in[2];
  const int*   Hh    = (const int*)d_in[3];
  const int*   raw   = (const int*)d_in[4];
  const int*   lgi   = (const int*)d_in[5];
  const float* Wt    = (const float*)d_in[6];
  const float* Wa    = (const float*)d_in[7];
  const float* a_src = (const float*)d_in[8];
  const float* a_dst = (const float*)d_in[9];
  const float* W_etn = (const float*)d_in[10];
  const float* W_edge= (const float*)d_in[11];
  const float* Ws_e0 = (const float*)d_in[12];
  const float* Wn_e0 = (const float*)d_in[13];
  const float* Ws_e1 = (const float*)d_in[14];
  const float* Wn_e1 = (const float*)d_in[15];
  const float* Ws_n0 = (const float*)d_in[16];
  const float* Wn_n0 = (const float*)d_in[17];
  const float* Ws_n1 = (const float*)d_in[18];
  const float* Wn_n1 = (const float*)d_in[19];
  const float* Ws_n2 = (const float*)d_in[20];
  const float* Wn_n2 = (const float*)d_in[21];
  const float* Wmix_n= (const float*)d_in[22];
  const float* Wmix_e= (const float*)d_in[23];
  const float* amix_n= (const float*)d_in[24];
  const float* amix_e= (const float*)d_in[25];
  const float* W_out = (const float*)d_in[26];
  const int *H0 = Hh, *H1 = Hh + EE;
  const int *rsrc = raw, *rdst = raw + EE;
  const int *lsrc = lgi, *ldst = lgi + LGE;

  float* ws = (float*)d_ws;
  size_t o = 0;
  auto take = [&](size_t n) { float* p = ws + o; o += n; return p; };
  float* f_h    = take((size_t)EE * HID);   // h, later tsae
  float* f_g    = take((size_t)EE * HID);   // gat accum, later ef
  float* f_hs   = take(EE);
  float* f_hd   = take(EE);
  float* f_mx   = take(EE);
  float* f_dn   = take(EE);                 // den -> 1/max(den,eps)
  float* f_nacc = take((size_t)NN * FIN);   // node accumulator (up to 256 wide)
  float* f_cnt  = take(NN);                 // edge-incidence counts
  float* f_deg  = take(NN);                 // raw-edge in-degree -> 1/max(deg,1)
  float* f_a    = take((size_t)NN * HID);
  float* f_b    = take((size_t)NN * HID);
  float* f_c    = take((size_t)NN * HID);
  float* f_d    = take((size_t)NN * HID);

  dim3 blk(256);
  const int FB = 2048;
  const int gE  = (EE + 127) / 128;   // gemm grid (edges)
  const int gN  = (NN + 127) / 128;   // gemm grid (nodes)
  const int wE  = (EE + 7) / 8;       // wave-per-edge grid
  const int wLG = (LGE + 7) / 8;
  const int wN  = (NN + 7) / 8;

  // ---- edge GCN ----
  // h = ea @ Wa
  gemm2_wmma<8><<<gE, blk, 0, stream>>>(ea, Wa, AD, nullptr, nullptr, 0, nullptr,
                                        nullptr, nullptr, f_h, EE, 0);
  edge_score_kernel<<<wE, blk, 0, stream>>>(f_h, a_src, a_dst, f_hs, f_hd, EE);
  fill_kernel<<<FB, blk, 0, stream>>>(f_mx, -3.0e38f, (long long)EE);
  fill_kernel<<<FB, blk, 0, stream>>>(f_dn, 0.f, (long long)EE);
  fill_kernel<<<FB, blk, 0, stream>>>(f_g, 0.f, (long long)EE * HID);
  gat_max_kernel<<<(LGE + 255) / 256, blk, 0, stream>>>(f_hs, f_hd, lsrc, ldst, f_mx, LGE);
  gat_accum_kernel<<<wLG, blk, 0, stream>>>(f_h, f_hs, f_hd, lsrc, ldst, f_mx, f_dn, f_g, LGE);
  recip_eps_kernel<<<FB, blk, 0, stream>>>(f_dn, (long long)EE);
  // tsae = lrelu(gat/den + et@Wt)   (overwrite f_h)
  gemm2_wmma<8><<<gE, blk, 0, stream>>>(et, Wt, TD, nullptr, nullptr, 0, nullptr,
                                        f_g, f_dn, f_h, EE, 2);
  // ef = tsae @ W_etn  -> f_g
  gemm2_wmma<8><<<gE, blk, 0, stream>>>(f_h, W_etn, HID, nullptr, nullptr, 0, nullptr,
                                        nullptr, nullptr, f_g, EE, 0);
  // EdgeToNode mean + lrelu
  fill_kernel<<<FB, blk, 0, stream>>>(f_nacc, 0.f, (long long)NN * HID);
  fill_kernel<<<FB, blk, 0, stream>>>(f_cnt, 0.f, (long long)NN);
  scatter_etn_kernel<<<wE, blk, 0, stream>>>(f_g, H0, H1, f_nacc, f_cnt, EE);
  etn_finalize_kernel<<<FB, blk, 0, stream>>>(f_nacc, f_cnt, f_a, (long long)NN * HID);
  // edge_repr = lrelu(node_e) @ W_edge  -> f_c
  gemm2_wmma<8><<<gN, blk, 0, stream>>>(f_a, W_edge, HID, nullptr, nullptr, 0, nullptr,
                                        nullptr, nullptr, f_c, NN, 0);

  // ---- raw-edge inverse degree (shared by all SAGE layers) ----
  fill_kernel<<<FB, blk, 0, stream>>>(f_deg, 0.f, (long long)NN);
  deg_count_kernel<<<(EE + 255) / 256, blk, 0, stream>>>(rdst, f_deg, EE);
  recip_max1_kernel<<<FB, blk, 0, stream>>>(f_deg, (long long)NN);

  // ---- SAGE over edge_repr (2 layers) ----
  fill_kernel<<<FB, blk, 0, stream>>>(f_nacc, 0.f, (long long)NN * HID);
  scatter_rows_kernel<<<wE, blk, 0, stream>>>(f_c, rsrc, rdst, f_nacc, EE, HID);
  gemm2_wmma<8><<<gN, blk, 0, stream>>>(f_c, Ws_e0, HID, f_nacc, Wn_e0, HID, f_deg,
                                        nullptr, nullptr, f_a, NN, 1);
  fill_kernel<<<FB, blk, 0, stream>>>(f_nacc, 0.f, (long long)NN * HID);
  scatter_rows_kernel<<<wE, blk, 0, stream>>>(f_a, rsrc, rdst, f_nacc, EE, HID);
  gemm2_wmma<8><<<gN, blk, 0, stream>>>(f_a, Ws_e1, HID, f_nacc, Wn_e1, HID, f_deg,
                                        nullptr, nullptr, f_b, NN, 0);  // aggr_edge

  // ---- SAGE over node features (3 layers) ----
  fill_kernel<<<FB, blk, 0, stream>>>(f_nacc, 0.f, (long long)NN * FIN);
  scatter_rows_kernel<<<wE, blk, 0, stream>>>(x, rsrc, rdst, f_nacc, EE, FIN);
  gemm2_wmma<8><<<gN, blk, 0, stream>>>(x, Ws_n0, FIN, f_nacc, Wn_n0, FIN, f_deg,
                                        nullptr, nullptr, f_c, NN, 1);
  fill_kernel<<<FB, blk, 0, stream>>>(f_nacc, 0.f, (long long)NN * HID);
  scatter_rows_kernel<<<wE, blk, 0, stream>>>(f_c, rsrc, rdst, f_nacc, EE, HID);
  gemm2_wmma<8><<<gN, blk, 0, stream>>>(f_c, Ws_n1, HID, f_nacc, Wn_n1, HID, f_deg,
                                        nullptr, nullptr, f_d, NN, 1);
  fill_kernel<<<FB, blk, 0, stream>>>(f_nacc, 0.f, (long long)NN * HID);
  scatter_rows_kernel<<<wE, blk, 0, stream>>>(f_d, rsrc, rdst, f_nacc, EE, HID);
  gemm2_wmma<8><<<gN, blk, 0, stream>>>(f_d, Ws_n2, HID, f_nacc, Wn_n2, HID, f_deg,
                                        nullptr, nullptr, f_a, NN, 0);  // node_repr

  // ---- MixAttention + classifier ----
  gemm2_wmma<8><<<gN, blk, 0, stream>>>(f_a, Wmix_n, HID, nullptr, nullptr, 0, nullptr,
                                        nullptr, nullptr, f_c, NN, 0);  // hn
  gemm2_wmma<8><<<gN, blk, 0, stream>>>(f_b, Wmix_e, HID, nullptr, nullptr, 0, nullptr,
                                        nullptr, nullptr, f_d, NN, 0);  // he
  mix_kernel<<<wN, blk, 0, stream>>>(f_c, f_d, amix_n, amix_e, f_a, NN);
  gemm2_wmma<4><<<gN, blk, 0, stream>>>(f_a, W_out, HID, nullptr, nullptr, 0, nullptr,
                                        nullptr, nullptr, (float*)d_out, NN, 0);
  logsoftmax_kernel<<<wN, blk, 0, stream>>>((float*)d_out, NN);
}
